// InterTransformerBlock_33775622816446
// MI455X (gfx1250) — matrix operand order repaired and checked by reading, not compiled, not run on hardware
//
#include <hip/hip_runtime.h>
#include <hip/hip_bf16.h>
#include <stdint.h>

typedef _Float16 half_t;
typedef __attribute__((ext_vector_type(16))) _Float16 v16h;
typedef __attribute__((ext_vector_type(8)))  _Float16 v8h;
typedef __attribute__((ext_vector_type(8)))  float    v8f;

// Problem constants (from reference)
#define B_      8
#define H_      128
#define W_      128
#define DIM_    256
#define CATT    128
#define NWIN    256      // (128/8)^2 windows per image
#define NELEM   64       // 8*8 elements per window
#define MT      131072   // B*H*W tokens (also B*NWIN*NELEM per branch)
#define HEADS   4
#define HD      32
#define NSLICE  1024     // B*HEADS*HD
#define HIDDEN_ 512

// ---------------------------------------------------------------------------
// f32 -> f16 conversion (weights)
// ---------------------------------------------------------------------------
__global__ void f32tof16_k(const float* __restrict__ s, half_t* __restrict__ d, int n) {
  int i = blockIdx.x * blockDim.x + threadIdx.x;
  if (i < n) d[i] = (half_t)s[i];
}

// ---------------------------------------------------------------------------
// Block-cooperative WMMA GEMM (K is a compile-time constant so the k-chunk
// loop fully unrolls and LDS/global offsets fold into immediates):
//   C[M,N] = A[M,K](f16,row) * W[N,K](f16,row)^T (+bias) (+= C if accum)
// Block = 8 waves = 128(M) x 64(N) super-tile; each wave owns a 16x64 strip.
// The shared 64x32 B tile is double-buffered in LDS, staged by wave 0 with
// gfx1250 async-to-LDS loads (ASYNCcnt) overlapped with WMMA on the other
// buffer; retired with s_wait_asynccnt + workgroup barrier.
// A/B fragments use contiguous k = 16*(lane/16) + s, not ISA table order; f32 C/D per ISA.
// ---------------------------------------------------------------------------
template <int K>
__global__ void gemm_f16_wmma(const half_t* __restrict__ A, const half_t* __restrict__ Wt,
                              const float* __restrict__ bias, float* __restrict__ C,
                              int M, int N, int accum) {
  __shared__ half_t Bsm[2][64 * 32];   // [buf][row(n) * 32 + k] halves, 4 KB each
  constexpr int CHUNKS = K >> 5;

  int wid  = threadIdx.x >> 5;         // 0..7
  int lane = threadIdx.x & 31;
  int nq = N >> 6;
  int bM = blockIdx.x / nq;
  int bN = blockIdx.x - bM * nq;
  int m0 = (bM * 8 + wid) << 4;
  int n0 = bN << 6;

  int rowA = m0 + (lane & 15);
  int kh = (lane >> 4) << 3;           // unused (0 or 8); A and B both use ks
  int ks = (lane >> 4) << 4;           // B: 0 or 16 halves within 32-K step
  int col = lane & 15;
  const half_t* Arow = A + (size_t)rowA * K;

  v8f acc[4];
#pragma unroll
  for (int q = 0; q < 4; ++q) {
    if (accum) {
#pragma unroll
      for (int r = 0; r < 8; ++r) {
        int rr = m0 + r + ((lane >> 4) << 3);
        int cc = n0 + (q << 4) + col;
        acc[q][r] = C[(size_t)rr * N + cc];
      }
    } else {
      v8f z = {};
      acc[q] = z;
    }
  }

  // --- prologue: stage chunk 0 into buffer 0 (wave 0 only), async then wait
  if (wid == 0) {
    const half_t* src = Wt + (size_t)n0 * K;
#pragma unroll
    for (int i = 0; i < 8; ++i) {
      int c = (i << 5) + lane;                // 0..255 16-byte packets
      int rowi = c >> 2, part = c & 3;
      const half_t* g = src + (size_t)rowi * K + (part << 3);
      unsigned ldsoff = (unsigned)(size_t)&Bsm[0][rowi * 32 + (part << 3)];
      asm volatile("global_load_async_to_lds_b128 %0, %1, off"
                   :: "v"(ldsoff), "v"(g) : "memory");
    }
    asm volatile("s_wait_asynccnt 0x0" ::: "memory");
  }
  __syncthreads();

#pragma unroll
  for (int kc = 0; kc < CHUNKS; ++kc) {
    const int cur = kc & 1;
    const bool pre = (kc + 1) < CHUNKS;

    // --- wave 0 kicks off the next B tile asynchronously (no wait yet)
    if (pre && wid == 0) {
      const half_t* src = Wt + (size_t)n0 * K + ((kc + 1) << 5);
#pragma unroll
      for (int i = 0; i < 8; ++i) {
        int c = (i << 5) + lane;
        int rowi = c >> 2, part = c & 3;
        const half_t* g = src + (size_t)rowi * K + (part << 3);
        unsigned ldsoff = (unsigned)(size_t)&Bsm[cur ^ 1][rowi * 32 + (part << 3)];
        asm volatile("global_load_async_to_lds_b128 %0, %1, off"
                     :: "v"(ldsoff), "v"(g) : "memory");
      }
    }
    if (pre) __builtin_prefetch((const void*)(Arow + ((kc + 1) << 5)), 0, 0);

    // --- compute on current buffer: batch all fragment loads, then 4 WMMAs
    const int k0 = kc << 5;
    v16h a = *(const v16h*)(Arow + k0 + ks);
    v16h bfrag[4];
#pragma unroll
    for (int q = 0; q < 4; ++q)
      bfrag[q] = *(const v16h*)&Bsm[cur][(((q << 4) + col) << 5) + ks];
#pragma unroll
    for (int q = 0; q < 4; ++q)
      acc[q] = __builtin_amdgcn_wmma_f32_16x16x32_f16(
          false, a, false, bfrag[q], (short)0, acc[q], false, false);

    // --- retire async stage, then block barrier to publish next buffer
    if (pre && wid == 0) {
      asm volatile("s_wait_asynccnt 0x0" ::: "memory");
    }
    __syncthreads();
  }

#pragma unroll
  for (int q = 0; q < 4; ++q) {
    int cc = n0 + (q << 4) + col;
    float bv = bias ? bias[cc] : 0.0f;
#pragma unroll
    for (int r = 0; r < 8; ++r) {
      int rr = m0 + r + ((lane >> 4) << 3);
      C[(size_t)rr * N + cc] = acc[q][r] + bv;
    }
  }
}

// ---------------------------------------------------------------------------
// LN1 + channel split + (-4,-4) roll of second half + window partition.
// One wave per token; lane c-range = [lane*8, lane*8+8). Writes f16 GEMM-A
// buffers for both attention branches.
// ---------------------------------------------------------------------------
__global__ void ln1_part_k(const float* __restrict__ x, const float* __restrict__ g,
                           const float* __restrict__ bb,
                           half_t* __restrict__ hA0, half_t* __restrict__ hA1) {
  int wave = blockIdx.x * (blockDim.x >> 5) + (threadIdx.x >> 5);
  int lane = threadIdx.x & 31;
  if (wave >= MT) return;
  int b  = wave >> 14;
  int y  = (wave >> 7) & 127;
  int xq = wave & 127;

  const float* xp = x + (size_t)wave * DIM_ + lane * 8;
  float v[8]; float s = 0.f, s2 = 0.f;
#pragma unroll
  for (int t = 0; t < 8; ++t) { v[t] = xp[t]; s += v[t]; s2 += v[t] * v[t]; }
#pragma unroll
  for (int off = 16; off > 0; off >>= 1) {
    s  += __shfl_xor(s,  off, 32);
    s2 += __shfl_xor(s2, off, 32);
  }
  float mean = s * (1.0f / DIM_);
  float var  = s2 * (1.0f / DIM_) - mean * mean;
  float rstd = rsqrtf(var + 1e-5f);

  int cbase = lane * 8;
  v8h o;
#pragma unroll
  for (int t = 0; t < 8; ++t)
    o[t] = (half_t)((v[t] - mean) * rstd * g[cbase + t] + bb[cbase + t]);

  if (lane < 16) {               // branch 0: channels 0..127, unshifted
    int win = (y >> 3) * 16 + (xq >> 3);
    int el  = (y & 7) * 8 + (xq & 7);
    size_t tok = ((size_t)b * NWIN + win) * NELEM + el;
    *(v8h*)(hA0 + tok * CATT + cbase) = o;
  } else {                       // branch 1: channels 128..255, rolled (-4,-4)
    int yt = (y - 4) & 127, xt = (xq - 4) & 127;
    int win = (yt >> 3) * 16 + (xt >> 3);
    int el  = (yt & 7) * 8 + (xt & 7);
    size_t tok = ((size_t)b * NWIN + win) * NELEM + el;
    *(v8h*)(hA1 + tok * CATT + (cbase - 128)) = o;
  }
}

// ---------------------------------------------------------------------------
// Grad-score: qg/kg[slice, win] from qkv f32 buffer (token-major, 384 ch).
// score = sum over 8x8 window of |dx| + |dy| with zero-prepend diffs.
// ---------------------------------------------------------------------------
__global__ void gradscore_k(const float* __restrict__ qkv,
                            float* __restrict__ qg, float* __restrict__ kg) {
  int tid = blockIdx.x * blockDim.x + threadIdx.x;
  if (tid >= NSLICE * NWIN) return;
  int win = tid & 255;
  int slice = tid >> 8;
  int b = slice >> 7, h = (slice >> 5) & 3, d = slice & 31;
  size_t tokbase = ((size_t)b * NWIN + win) * NELEM;

#pragma unroll
  for (int s = 0; s < 2; ++s) {
    int ch = s * CATT + h * HD + d;
    float scale = (s == 0) ? 0.17677669529663687f : 1.0f;  // hd^-0.5 for q
    float prow[8]; float score = 0.f;
    for (int y = 0; y < 8; ++y) {
      float pvx = 0.f;
#pragma unroll
      for (int xx = 0; xx < 8; ++xx) {
        float v = qkv[(tokbase + y * 8 + xx) * 384 + ch] * scale;
        float up = (y > 0) ? prow[xx] : 0.f;
        score += fabsf(v - pvx) + fabsf(v - up);
        prow[xx] = v; pvx = v;
      }
    }
    float* dst = (s == 0) ? qg : kg;
    dst[(size_t)slice * NWIN + win] = score;
  }
}

// ---------------------------------------------------------------------------
// Softmax stats: row max and 1/denominator for s_ij = qg_i * kg_j over j.
// ---------------------------------------------------------------------------
__global__ void softstat_k(const float* __restrict__ qg, const float* __restrict__ kg,
                           float* __restrict__ mrow, float* __restrict__ dinv) {
  int tid = blockIdx.x * blockDim.x + threadIdx.x;
  if (tid >= NSLICE * NWIN) return;
  int i = tid & 255;
  int slice = tid >> 8;
  const float* kgs = kg + (size_t)slice * NWIN;
  float qv = qg[(size_t)slice * NWIN + i];
  float mx = -3.4e38f;
  for (int j = 0; j < NWIN; ++j) mx = fmaxf(mx, qv * kgs[j]);
  float s = 0.f;
  for (int j = 0; j < NWIN; ++j) s += __expf(qv * kgs[j] - mx);
  mrow[tid] = mx;
  dinv[tid] = 1.0f / s;
}

// ---------------------------------------------------------------------------
// Reorganize V into per-slice transposed f16: Vt[slice][ne(64)][j(256)].
// ---------------------------------------------------------------------------
__global__ void vreorg_k(const float* __restrict__ qkv, half_t* __restrict__ Vt) {
  int tid = blockIdx.x * blockDim.x + threadIdx.x;
  if (tid >= NSLICE * NELEM * NWIN) return;
  int j  = tid & 255;
  int ne = (tid >> 8) & 63;
  int slice = tid >> 14;
  int b = slice >> 7, h = (slice >> 5) & 3, d = slice & 31;
  float v = qkv[(((size_t)b * NWIN + j) * NELEM + ne) * 384 + 256 + h * HD + d];
  Vt[(size_t)slice * (NELEM * NWIN) + (size_t)ne * NWIN + j] = (half_t)v;
}

// ---------------------------------------------------------------------------
// Attention einsum via WMMA: per (slice, itile) wave computes a 16(i) x 64(N)
// tile of out = softmax(qg_i*kg_j) @ V. Probabilities are built on the fly
// into the A-fragment; V B-fragments are contiguous loads from Vt. Fully
// unrolled k-loop (8 x 32) -> 32 WMMAs pipelined against the exp VALU work.
// Output stored f16 token-major as the proj GEMM A matrix.
// ---------------------------------------------------------------------------
__global__ void attn_wmma_k(const float* __restrict__ qg, const float* __restrict__ kg,
                            const float* __restrict__ mrow, const float* __restrict__ dinv,
                            const half_t* __restrict__ Vt, half_t* __restrict__ projA) {
  int wave = blockIdx.x * (blockDim.x >> 5) + (threadIdx.x >> 5);
  int lane = threadIdx.x & 31;
  int slice = wave >> 4;
  int itile = wave & 15;
  if (slice >= NSLICE) return;
  int b = slice >> 7, h = (slice >> 5) & 3, d = slice & 31;
  size_t sbase = (size_t)slice * NWIN;

  int i0 = itile << 4;
  int row = i0 + (lane & 15);
  float qv = qg[sbase + row];
  float mv = mrow[sbase + row];
  float di = dinv[sbase + row];
  int kh = (lane >> 4) << 3;
  int ks = (lane >> 4) << 4;
  const half_t* VtS = Vt + sbase * NELEM;  // slice * 64 * 256 halves
  int col = lane & 15;

  v8f acc[4];
#pragma unroll
  for (int q = 0; q < 4; ++q) { v8f z = {}; acc[q] = z; }

#pragma unroll
  for (int kc = 0; kc < 8; ++kc) {
    const int k0 = kc << 5;
    v16h a;
#pragma unroll
    for (int t = 0; t < 8; ++t) {
      float k0v = kg[sbase + k0 + ks + t];
      float k1v = kg[sbase + k0 + ks + 8 + t];
      a[t]     = (half_t)(__expf(qv * k0v - mv) * di);
      a[8 + t] = (half_t)(__expf(qv * k1v - mv) * di);
    }
    v16h bfrag[4];
#pragma unroll
    for (int q = 0; q < 4; ++q)
      bfrag[q] = *(const v16h*)(VtS + (size_t)((q << 4) + col) * NWIN + k0 + ks);
#pragma unroll
    for (int q = 0; q < 4; ++q)
      acc[q] = __builtin_amdgcn_wmma_f32_16x16x32_f16(
          false, a, false, bfrag[q], (short)0, acc[q], false, false);
  }

  int ch = h * HD + d;
#pragma unroll
  for (int q = 0; q < 4; ++q) {
#pragma unroll
    for (int r = 0; r < 8; ++r) {
      int i  = i0 + r + ((lane >> 4) << 3);
      int ne = (q << 4) + col;
      size_t token = ((size_t)b * NWIN + i) * NELEM + ne;
      projA[token * CATT + ch] = (half_t)acc[q][r];
    }
  }
}

// ---------------------------------------------------------------------------
// Window reverse + roll(+4,+4) of branch1 + concat + residual: out = x + xa.
// ---------------------------------------------------------------------------
__global__ void resid_k(const float* __restrict__ x, const float* __restrict__ a0,
                        const float* __restrict__ a1, float* __restrict__ out) {
  long long tid = (long long)blockIdx.x * blockDim.x + threadIdx.x;
  if (tid >= (long long)MT * DIM_) return;
  int c = (int)(tid & 255);
  long long t = tid >> 8;
  int b  = (int)(t >> 14);
  int y  = (int)(t >> 7) & 127;
  int xq = (int)t & 127;
  float av;
  if (c < 128) {
    int win = (y >> 3) * 16 + (xq >> 3);
    int el  = (y & 7) * 8 + (xq & 7);
    av = a0[(((size_t)b * NWIN + win) * NELEM + el) * CATT + c];
  } else {
    int ys = (y - 4) & 127, xs = (xq - 4) & 127;
    int win = (ys >> 3) * 16 + (xs >> 3);
    int el  = (ys & 7) * 8 + (xs & 7);
    av = a1[(((size_t)b * NWIN + win) * NELEM + el) * CATT + (c - 128)];
  }
  out[tid] = x[tid] + av;
}

// ---------------------------------------------------------------------------
// LN2: wave per token, writes f16 [token][256] GEMM-A buffer.
// ---------------------------------------------------------------------------
__global__ void ln2_k(const float* __restrict__ x, const float* __restrict__ g,
                      const float* __restrict__ bb, half_t* __restrict__ out) {
  int wave = blockIdx.x * (blockDim.x >> 5) + (threadIdx.x >> 5);
  int lane = threadIdx.x & 31;
  if (wave >= MT) return;
  const float* xp = x + (size_t)wave * DIM_ + lane * 8;
  float v[8]; float s = 0.f, s2 = 0.f;
#pragma unroll
  for (int t = 0; t < 8; ++t) { v[t] = xp[t]; s += v[t]; s2 += v[t] * v[t]; }
#pragma unroll
  for (int off = 16; off > 0; off >>= 1) {
    s  += __shfl_xor(s,  off, 32);
    s2 += __shfl_xor(s2, off, 32);
  }
  float mean = s * (1.0f / DIM_);
  float var  = s2 * (1.0f / DIM_) - mean * mean;
  float rstd = rsqrtf(var + 1e-5f);
  int cbase = lane * 8;
  v8h o;
#pragma unroll
  for (int t = 0; t < 8; ++t)
    o[t] = (half_t)((v[t] - mean) * rstd * g[cbase + t] + bb[cbase + t]);
  *(v8h*)(out + (size_t)wave * DIM_ + cbase) = o;
}

// ---------------------------------------------------------------------------
// Depthwise 3x3 conv (zero pad) + gate: gA[t][c] = (conv(c)+b)(conv(c+256)+b).
// fc1 output is token-major [token][512] f32.
// ---------------------------------------------------------------------------
__global__ void dwgate_k(const float* __restrict__ f1, const float* __restrict__ w,
                         const float* __restrict__ bias, half_t* __restrict__ gA) {
  long long tid = (long long)blockIdx.x * blockDim.x + threadIdx.x;
  if (tid >= (long long)MT * 256) return;
  int c = (int)(tid & 255);
  long long t = tid >> 8;
  int b  = (int)(t >> 14);
  int y  = (int)(t >> 7) & 127;
  int xx = (int)t & 127;
  float acc0 = bias[c], acc1 = bias[c + 256];
#pragma unroll
  for (int dy = -1; dy <= 1; ++dy) {
    int yy = y + dy;
    if (yy < 0 || yy > 127) continue;
#pragma unroll
    for (int dx = -1; dx <= 1; ++dx) {
      int xw = xx + dx;
      if (xw < 0 || xw > 127) continue;
      size_t tok = ((size_t)b * H_ + yy) * W_ + xw;
      const float* p = f1 + tok * HIDDEN_;
      int wi = (dy + 1) * 3 + (dx + 1);
      acc0 += w[c * 9 + wi] * p[c];
      acc1 += w[(c + 256) * 9 + wi] * p[c + 256];
    }
  }
  gA[(size_t)t * 256 + c] = (half_t)(acc0 * acc1);
}

// ---------------------------------------------------------------------------
// Host-side launcher
// ---------------------------------------------------------------------------
static inline int cdiv(long long a, long long b) { return (int)((a + b - 1) / b); }

extern "C" void kernel_launch(void* const* d_in, const int* in_sizes, int n_in,
                              void* d_out, int out_size, void* d_ws, size_t ws_size,
                              hipStream_t stream) {
  const float* x       = (const float*)d_in[0];
  const float* n1g     = (const float*)d_in[1];
  const float* n1b     = (const float*)d_in[2];
  const float* qkv_w   = (const float*)d_in[3];
  const float* qkv_b   = (const float*)d_in[4];
  const float* proj_w  = (const float*)d_in[5];
  const float* proj_b  = (const float*)d_in[6];
  const float* n2g     = (const float*)d_in[7];
  const float* n2b     = (const float*)d_in[8];
  const float* fc1_w   = (const float*)d_in[9];
  const float* fc1_b   = (const float*)d_in[10];
  const float* dw_w    = (const float*)d_in[11];
  const float* dw_b    = (const float*)d_in[12];
  const float* fc2_w   = (const float*)d_in[13];
  const float* fc2_b   = (const float*)d_in[14];
  float* out = (float*)d_out;

  char* ws = (char*)d_ws;
  const size_t MB = 1ull << 20;
  // workspace layout (bytes)
  half_t* hW    = (half_t*)(ws + 0);                 //  1 MB  f16 weights
  half_t* wQKV  = hW;                                //  49152 halves
  half_t* wPROJ = hW + 65536;                        //  16384
  half_t* wFC1  = hW + 98304;                        // 131072
  half_t* wFC2  = hW + 262144;                       //  65536
  float*  QG    = (float*)(ws + 1 * MB);             //  2 MB (2 branches)
  float*  KG    = (float*)(ws + 3 * MB);             //  2 MB
  float*  MM    = (float*)(ws + 5 * MB);             //  2 MB
  float*  DI    = (float*)(ws + 7 * MB);             //  2 MB
  half_t* HA    = (half_t*)(ws + 9 * MB);            // 64 MB (LN1 A / LN2 A / gate A)
  half_t* HVT   = (half_t*)(ws + 73 * MB);           // 64 MB (Vt, 2 branches)
  half_t* PROJA = (half_t*)(ws + 137 * MB);          // 64 MB (attn out f16, 2 branches)
  float*  AOUT  = (float*)(ws + 201 * MB);           // 128 MB (proj out f32, 2 branches)
  float*  S1    = (float*)(ws + 329 * MB);           // 256 MB (qkv f32 / fc1 out f32)

  const int TB = 256;

  // --- convert weights to f16
  f32tof16_k<<<cdiv(3 * CATT * CATT, TB), TB, 0, stream>>>(qkv_w,  wQKV,  3 * CATT * CATT);
  f32tof16_k<<<cdiv(CATT * CATT, TB),     TB, 0, stream>>>(proj_w, wPROJ, CATT * CATT);
  f32tof16_k<<<cdiv(HIDDEN_ * DIM_, TB),  TB, 0, stream>>>(fc1_w,  wFC1,  HIDDEN_ * DIM_);
  f32tof16_k<<<cdiv(DIM_ * DIM_, TB),     TB, 0, stream>>>(fc2_w,  wFC2,  DIM_ * DIM_);

  // --- LN1 + split + shift + window partition -> f16 A for both branches
  half_t* hA0 = HA;
  half_t* hA1 = HA + (size_t)MT * CATT;
  ln1_part_k<<<MT / 8, TB, 0, stream>>>(x, n1g, n1b, hA0, hA1);

  // --- per-branch attention pipeline
  for (int br = 0; br < 2; ++br) {
    half_t* hA   = (br == 0) ? hA0 : hA1;
    float*  qg   = QG + (size_t)br * NSLICE * NWIN;
    float*  kg   = KG + (size_t)br * NSLICE * NWIN;
    float*  mm   = MM + (size_t)br * NSLICE * NWIN;
    float*  di   = DI + (size_t)br * NSLICE * NWIN;
    half_t* vt   = HVT   + (size_t)br * NSLICE * NELEM * NWIN;
    half_t* pA   = PROJA + (size_t)br * MT * CATT;
    float*  aout = AOUT  + (size_t)br * MT * CATT;

    // qkv GEMM: [131072 x 128] x [384 x 128]^T -> S1
    gemm_f16_wmma<CATT><<<(MT / 128) * (384 / 64), TB, 0, stream>>>(
        hA, wQKV, qkv_b, S1, MT, 384, 0);
    gradscore_k<<<cdiv(NSLICE * NWIN, TB), TB, 0, stream>>>(S1, qg, kg);
    softstat_k<<<cdiv(NSLICE * NWIN, TB), TB, 0, stream>>>(qg, kg, mm, di);
    vreorg_k<<<cdiv((long long)NSLICE * NELEM * NWIN, TB), TB, 0, stream>>>(S1, vt);
    // attention: 1024 slices x 16 i-tiles waves
    attn_wmma_k<<<cdiv((long long)NSLICE * 16, 8), TB, 0, stream>>>(qg, kg, mm, di, vt, pA);
    // proj GEMM: [131072 x 128] x [128 x 128]^T -> aout
    gemm_f16_wmma<CATT><<<(MT / 128) * (CATT / 64), TB, 0, stream>>>(
        pA, wPROJ, proj_b, aout, MT, CATT, 0);
  }

  // --- window reverse + roll + concat + residual -> d_out (xres)
  resid_k<<<cdiv((long long)MT * DIM_, TB), TB, 0, stream>>>(
      x, AOUT, AOUT + (size_t)MT * CATT, out);

  // --- LN2 -> f16 A
  ln2_k<<<MT / 8, TB, 0, stream>>>(out, n2g, n2b, HA);

  // --- fc1 GEMM: [131072 x 256] x [512 x 256]^T -> S1
  gemm_f16_wmma<DIM_><<<(MT / 128) * (HIDDEN_ / 64), TB, 0, stream>>>(
      HA, wFC1, fc1_b, S1, MT, HIDDEN_, 0);

  // --- depthwise conv + gate -> f16 gate A (reuse HA)
  dwgate_k<<<cdiv((long long)MT * 256, TB), TB, 0, stream>>>(S1, dw_w, dw_b, HA);

  // --- fc2 GEMM with accumulate into d_out: out = xres + gate @ fc2^T + b
  gemm_f16_wmma<DIM_><<<(MT / 128) * (DIM_ / 64), TB, 0, stream>>>(
      HA, wFC2, fc2_b, out, MT, DIM_, 1);
}
